// TS_CoherAnalysis_54958401519616
// MI455X (gfx1250) — matrix operand order repaired and checked by reading, not compiled, not run on hardware
//
#include <hip/hip_runtime.h>
#include <hip/hip_bf16.h>

#define B_      8
#define CT      16
#define LL      2048
#define NN      2048
#define PSEG    128
#define STEP    64
#define SEGS    31
#define FREQ    65
#define FTILES  5
#define NTILES  128
#define KREF    32

typedef __attribute__((ext_vector_type(16))) __bf16 v16bf;
typedef __attribute__((ext_vector_type(8)))  float  v8f;

// Native f32 -> bf16 conversion (RNE); backend lowers to v_cvt_*bf16_f32.
__device__ __forceinline__ unsigned short f2bf(float x) {
    __bf16 h = (__bf16)x;
    return __builtin_bit_cast(unsigned short, h);
}

__device__ __forceinline__ unsigned pk2bf(float lo, float hi) {
    return (unsigned)f2bf(lo) | ((unsigned)f2bf(hi) << 16);
}

union BfTile { v16bf v; unsigned short u[16]; };

// ---------------------------------------------------------------------------
// Kernel 0: bake DFT operand tiles in WMMA B-matrix (32x16 bf16) layout.
// tile index = (kk*5 + ftile)*2 + isSin ; each tile = 1 KB (32 lanes x 32 B).
// B layout: lane l holds column n=l&15, rows k=(j>>2)*16 + (l>>4)*8 + (j&3)*2 + p.
// ---------------------------------------------------------------------------
__global__ void __launch_bounds__(32) dft_table_kernel(unsigned char* __restrict__ tab) {
    int tile = blockIdx.x;            // 0..39
    int lane = threadIdx.x;
    int kk = tile / 10, rem = tile % 10;
    int ft = rem >> 1, isSin = rem & 1;
    int f  = ft * 16 + (lane & 15);
    unsigned short u[16];
#pragma unroll
    for (int j = 0; j < 8; ++j)
#pragma unroll
        for (int p = 0; p < 2; ++p) {
            int k = ((j >> 2) << 4) + ((lane >> 4) << 3) + ((j & 3) << 1) + p;   // row in 32x16 tile
            int t = kk * 32 + k;                          // time sample 0..127
            float val = 0.f;
            if (f < FREQ) {
                int ph = (f * t) & 127;                   // exact phase reduction
                float ang = (float)ph * (6.28318530718f / 128.f);
                val = isSin ? -__sinf(ang) : __cosf(ang); // conj sign baked in
            }
            u[j * 2 + p] = f2bf(val);
        }
    unsigned* dst = (unsigned*)(tab + (size_t)tile * 1024 + lane * 32);
#pragma unroll
    for (int i = 0; i < 8; ++i) dst[i] = ((unsigned)u[2 * i + 1] << 16) | u[2 * i];
}

// ---------------------------------------------------------------------------
// Kernel 1: Welch spectra via bf16 WMMA DFT. One wave per series row.
// A = [16 segs x 128 samples] bf16 (4 K-steps); B = baked DFT tiles.
// Emits: packed bf16 spectra tiles (A-layout for target, B-layout for db)
// via single b128 stores per (mt, re/im), and raw power sums Sxx / Syy.
// ---------------------------------------------------------------------------
__global__ void __launch_bounds__(32)
welch_dft_kernel(const float* __restrict__ src, int rows_per_b, int is_target,
                 const unsigned char* __restrict__ tab,
                 unsigned char* __restrict__ xpack,
                 unsigned char* __restrict__ ypack,
                 float* __restrict__ sxx, float* __restrict__ syy) {
    __shared__ float lx[LL];
    __shared__ float lmean[32];
    __shared__ float lwin[PSEG];

    int lane = threadIdx.x;
    int row  = blockIdx.x;
    int b    = row / rows_per_b;
    int r    = row - b * rows_per_b;

    const float4* s4 = (const float4*)(src + (size_t)row * LL);
    for (int i = lane; i < LL / 4; i += 32) ((float4*)lx)[i] = s4[i];
    for (int t = lane; t < PSEG; t += 32)
        lwin[t] = 0.5f - 0.5f * __cosf((float)t * (6.28318530718f / 128.f));
    __syncthreads();

    // per-segment means (detrend='constant')
    {
        float acc = 0.f;
        if (lane < SEGS) {
            int base = lane * STEP;
            for (int t = 0; t < PSEG; ++t) acc += lx[base + t];
        }
        lmean[lane] = acc * (1.f / 128.f);
    }
    __syncthreads();

    // Build all 8 A operands (2 seg-tiles x 4 K-steps) in wave32 A-layout:
    // lane l: M = l&15, K pair in vgpr j: k = (j>=4)*16 + (l>=16)*8 + (j&3)*2 + p
    BfTile A[2][4];
    int m = lane & 15, kh = lane >> 4;
#pragma unroll
    for (int mt = 0; mt < 2; ++mt) {
        int seg = mt * 16 + m;
        bool valid = seg < SEGS;              // seg 31 = zero padding
        float mu = valid ? lmean[seg] : 0.f;
        int base = seg * STEP;
#pragma unroll
        for (int kk = 0; kk < 4; ++kk)
#pragma unroll
            for (int j = 0; j < 8; ++j)
#pragma unroll
                for (int p = 0; p < 2; ++p) {
                    int tl = ((j >> 2) << 4) + (kh << 3) + ((j & 3) << 1) + p;
                    int t  = kk * 32 + tl;
                    float val = valid ? (lx[base + t] - mu) * lwin[t] : 0.f;
                    A[mt][kk].u[j * 2 + p] = f2bf(val);
                }
    }

    int fl = lane & 15;
    for (int ft = 0; ft < FTILES; ++ft) {
        v16bf bc[4], bs[4];
#pragma unroll
        for (int kk = 0; kk < 4; ++kk) {
            const unsigned char* tb = tab + (size_t)((kk * 5 + ft) * 2) * 1024 + lane * 32;
            bc[kk] = *(const v16bf*)tb;
            bs[kk] = *(const v16bf*)(tb + 1024);
        }
        int f = ft * 16 + fl;
        float psum = 0.f;
#pragma unroll
        for (int mt = 0; mt < 2; ++mt) {
            v8f ar = {0.f,0.f,0.f,0.f,0.f,0.f,0.f,0.f};
            v8f ai = {0.f,0.f,0.f,0.f,0.f,0.f,0.f,0.f};
#pragma unroll
            for (int kk = 0; kk < 4; ++kk) {
                ar = __builtin_amdgcn_wmma_f32_16x16x32_bf16(false, A[mt][kk].v, false, bc[kk],
                                                             (short)0, ar, false, false);
                ai = __builtin_amdgcn_wmma_f32_16x16x32_bf16(false, A[mt][kk].v, false, bs[kk],
                                                             (short)0, ai, false, false);
            }
            // C/D layout: lane l, vgpr v: seg = mt*16 + v + 8*(l>>4), f = ft*16 + (l&15)
#pragma unroll
            for (int v = 0; v < 8; ++v) {
                float re = ar[v], im = ai[v];
                psum += re * re + im * im;
            }
            if (f < FREQ) {
                // The 8 column values v=0..7 are k-consecutive in both packed
                // layouts -> pack into one uint4 per (re/im) and b128-store.
                uint4 pr, pi;
                pr.x = pk2bf(ar[0], ar[1]); pr.y = pk2bf(ar[2], ar[3]);
                pr.z = pk2bf(ar[4], ar[5]); pr.w = pk2bf(ar[6], ar[7]);
                pi.x = pk2bf(ai[0], ai[1]); pi.y = pk2bf(ai[2], ai[3]);
                pi.z = pk2bf(ai[4], ai[5]); pi.w = pk2bf(ai[6], ai[7]);
                if (is_target) {
                    // A-layout tile for X: laneP = c + 16*kh, byte off = 16*mt + 2*v
                    size_t tb2 = ((size_t)(b * FREQ + f) * 2) * 1024;
                    int off = ((r & 15) + (kh << 4)) * 32 + (mt << 4);
                    *(uint4*)(xpack + tb2 + off)        = pr;
                    *(uint4*)(xpack + tb2 + 1024 + off) = pi;
                } else {
                    // B-layout tile for Y: laneP = n + 16*kh, byte off = 16*mt + 2*v
                    size_t tb2 = ((size_t)((b * FREQ + f) * NTILES + (r >> 4)) * 2) * 1024;
                    int off = ((r & 15) + (kh << 4)) * 32 + (mt << 4);
                    *(uint4*)(ypack + tb2 + off)        = pr;
                    *(uint4*)(ypack + tb2 + 1024 + off) = pi;
                }
            }
        }
        // combine the two seg-halves (lanes l and l^16 cover all 32 segs for same f)
        psum += __shfl_xor(psum, 16);
        if (lane < 16) {
            int f2 = ft * 16 + lane;
            if (f2 < FREQ) {
                if (is_target) sxx[(size_t)(b * FREQ + f2) * CT + r] = psum;
                else           syy[(size_t)(b * FREQ + f2) * NN + r] = psum;
            }
        }
    }
}

// ---------------------------------------------------------------------------
// Kernel 2: coherence scores. One wave per (b, n-tile of 16).
// Per f: 4 bf16 WMMAs (K=32 = padded S) -> Pxy_re/im 16x16 tile; accumulate
// |Pxy|^2 / (Sxx*Syy + S^2*eps) over f.
// ---------------------------------------------------------------------------
__global__ void __launch_bounds__(32)
score_kernel(const unsigned char* __restrict__ xpack,
             const unsigned char* __restrict__ ypack,
             const float* __restrict__ sxx, const float* __restrict__ syy,
             float* __restrict__ scores) {
    int lane = threadIdx.x;
    int nt = blockIdx.x & (NTILES - 1);
    int b  = blockIdx.x >> 7;
    float acc[8];
#pragma unroll
    for (int v = 0; v < 8; ++v) acc[v] = 0.f;
    int hb = (lane >> 4) << 3;

    for (int f = 0; f < FREQ; ++f) {
        const unsigned char* xb = xpack + ((size_t)(b * FREQ + f) * 2) * 1024 + lane * 32;
        v16bf xr = *(const v16bf*)xb;
        v16bf xi = *(const v16bf*)(xb + 1024);
        const unsigned char* yb =
            ypack + ((size_t)((b * FREQ + f) * NTILES + nt) * 2) * 1024 + lane * 32;
        v16bf yr = *(const v16bf*)yb;
        v16bf yi = *(const v16bf*)(yb + 1024);
        // prefetch next f's Y tile (streams 136 MB; emits global_prefetch_b8)
        if (f + 1 < FREQ)
            __builtin_prefetch(yb + (size_t)NTILES * 2 * 1024, 0, 0);

        v8f z = {0.f,0.f,0.f,0.f,0.f,0.f,0.f,0.f};
        v8f pre = __builtin_amdgcn_wmma_f32_16x16x32_bf16(false, xi, false, yi, (short)0, z, false, false);
        pre     = __builtin_amdgcn_wmma_f32_16x16x32_bf16(false, xr, false, yr, (short)0, pre, false, false);
        v8f t1  = __builtin_amdgcn_wmma_f32_16x16x32_bf16(false, xi, false, yr, (short)0, z, false, false);
        v8f t2  = __builtin_amdgcn_wmma_f32_16x16x32_bf16(false, xr, false, yi, (short)0, z, false, false);

        float pxl = sxx[(size_t)(b * FREQ + f) * CT + (lane & 15)];
        float pyl = syy[(size_t)(b * FREQ + f) * NN + nt * 16 + (lane & 15)];
#pragma unroll
        for (int v = 0; v < 8; ++v) {
            float px = __shfl(pxl, v + hb);           // Sxx for row m = v + 8*(lane>>4)
            float re = pre[v];
            float im = t1[v] - t2[v];                 // conj(Y): im = Xi*Yr - Xr*Yi
            acc[v] += (re * re + im * im) / (px * pyl + 9.61e-10f);
        }
    }
#pragma unroll
    for (int v = 0; v < 8; ++v) {
        int mrow = v + hb;
        int n = nt * 16 + (lane & 15);
        scores[((size_t)(b * CT + mrow)) * NN + n] = acc[v] * (1.f / 65.f);
    }
}

// ---------------------------------------------------------------------------
// Kernel 3: top-32 (descending, lowest-index tie-break like lax.top_k) + gather.
// One 256-thread block per (b, c).
// ---------------------------------------------------------------------------
__global__ void __launch_bounds__(256)
topk_gather_kernel(const float* __restrict__ scores, const float* __restrict__ db,
                   float* __restrict__ out) {
    __shared__ float vals[NN];
    __shared__ float rv[256];
    __shared__ int   ri[256];
    __shared__ int   topidx[KREF];

    int tid = threadIdx.x;
    int b = blockIdx.x >> 4;
    int c = blockIdx.x & 15;
    const float* srow = scores + ((size_t)(b * CT + c)) * NN;
    for (int i = tid; i < NN; i += 256) vals[i] = srow[i];
    __syncthreads();

    for (int k = 0; k < KREF; ++k) {
        float bv = -1e30f; int bi = NN;
        for (int i = tid; i < NN; i += 256) {
            float v = vals[i];
            if (v > bv || (v == bv && i < bi)) { bv = v; bi = i; }
        }
        rv[tid] = bv; ri[tid] = bi;
        __syncthreads();
        for (int off = 128; off > 0; off >>= 1) {
            if (tid < off) {
                float v2 = rv[tid + off]; int i2 = ri[tid + off];
                if (v2 > rv[tid] || (v2 == rv[tid] && i2 < ri[tid])) { rv[tid] = v2; ri[tid] = i2; }
            }
            __syncthreads();
        }
        if (tid == 0) { topidx[k] = ri[0]; vals[ri[0]] = -1e38f; }
        __syncthreads();
    }

    // gather selected rows: out[b, c*32 + k, :] = db[b, idx, :]
    for (int k = 0; k < KREF; ++k) {
        int idx = topidx[k];
        const float4* s = (const float4*)(db + ((size_t)b * NN + idx) * LL);
        float4* d = (float4*)(out + (((size_t)(b * CT + c)) * KREF + k) * LL);
        for (int i = tid; i < LL / 4; i += 256) d[i] = s[i];
    }
}

// ---------------------------------------------------------------------------
extern "C" void kernel_launch(void* const* d_in, const int* in_sizes, int n_in,
                              void* d_out, int out_size, void* d_ws, size_t ws_size,
                              hipStream_t stream) {
    const float* target = (const float*)d_in[0];   // [8,16,2048]
    const float* db     = (const float*)d_in[1];   // [8,2048,2048]
    float* out          = (float*)d_out;           // [8,512,2048]

    unsigned char* w      = (unsigned char*)d_ws;
    unsigned char* tab    = w;                                              // 40 KB
    unsigned char* xpack  = w + 40960;                                      // 1.04 MB
    unsigned char* ypack  = xpack + (size_t)B_ * FREQ * 2 * 1024;           // 136.3 MB
    float* sxx    = (float*)(ypack + (size_t)B_ * FREQ * NTILES * 2 * 1024);
    float* syy    = sxx + (size_t)B_ * FREQ * CT;
    float* scores = syy + (size_t)B_ * FREQ * NN;

    dft_table_kernel<<<dim3(40), dim3(32), 0, stream>>>(tab);
    welch_dft_kernel<<<dim3(B_ * CT), dim3(32), 0, stream>>>(
        target, CT, 1, tab, xpack, ypack, sxx, syy);
    welch_dft_kernel<<<dim3(B_ * NN), dim3(32), 0, stream>>>(
        db, NN, 0, tab, xpack, ypack, sxx, syy);
    score_kernel<<<dim3(B_ * NTILES), dim3(32), 0, stream>>>(
        xpack, ypack, sxx, syy, scores);
    topk_gather_kernel<<<dim3(B_ * CT), dim3(256), 0, stream>>>(scores, db, out);
}
